// RLAgent_28870770163721
// MI455X (gfx1250) — matrix operand order repaired and checked by reading, not compiled, not run on hardware
//
#include <hip/hip_runtime.h>
#include <hip/hip_bf16.h>
#include <cstdint>

typedef __attribute__((ext_vector_type(16))) _Float16 v16h;
typedef __attribute__((ext_vector_type(8)))  _Float16 v8h;
typedef __attribute__((ext_vector_type(8)))  float    v8f;

#define NNODES 50000
#define NEDGES 800000
#define NTOT   (NNODES + NEDGES)   // edges including self-loops = 850000

// ---------------------------------------------------------------- utilities

__global__ void k_f32_to_f16(const float* __restrict__ src, _Float16* __restrict__ dst, int n) {
  int i = blockIdx.x * blockDim.x + threadIdx.x;
  if (i < n) dst[i] = (_Float16)src[i];
}

// w: [K][Nc] row-major f32  ->  wt: [Nc][K] row-major f16
__global__ void k_transpose_f16(const float* __restrict__ w, _Float16* __restrict__ wt,
                                int K, int Nc) {
  int i = blockIdx.x * blockDim.x + threadIdx.x;
  if (i >= K * Nc) return;
  int k = i / Nc, c = i % Nc;
  wt[(size_t)c * K + k] = (_Float16)w[i];
}

__global__ void k_fill(float* __restrict__ p, float v, int n) {
  int i = blockIdx.x * blockDim.x + threadIdx.x;
  if (i < n) p[i] = v;
}

__device__ __forceinline__ void atomicMaxFloat(float* addr, float v) {
  // sign-aware trick: positive floats order as ints, negative floats order
  // inversely as uints. Works with -1e30 init.
  if (v >= 0.0f) atomicMax((int*)addr, __float_as_int(v));
  else           atomicMin((unsigned int*)addr, __float_as_uint(v));
}

// ---------------------------------------------------------------- WMMA GEMM
// C[M][Nc] (f32) = A[M][K] (f16, row-major) * B[K][Nc] via Bt[Nc][K] (f16).
// grid.x = M/16 row tiles; blockDim.x = 32 * (Nc/16); one wave per 16x16 tile.
__global__ void k_gemm_wmma(const _Float16* __restrict__ A,
                            const _Float16* __restrict__ Bt,
                            float* __restrict__ C, int K, int Nc) {
  const int lane = threadIdx.x & 31;
  const int wave = threadIdx.x >> 5;
  const int hg   = lane >> 4;     // lane half-group
  const int r    = lane & 15;
  const int rowBase = blockIdx.x * 16;
  const int colBase = wave * 16;

  v8f acc = {};
  // A 16-bit layout: lanes 0-15 hold K = {0..7, 16..23}; lanes 16-31 hold
  // K = {8..15, 24..31} (each +kb). Two contiguous 16B loads per fragment.
  const _Float16* arow = A  + (size_t)(rowBase + r) * K + 8 * hg;
  // B 16-bit layout: lane n = column N, lanes 0-15 K = {0..7, 16..23}, lanes 16-31 K = {8..15, 24..31}.
  const _Float16* brow = Bt + (size_t)(colBase + r) * K + 8 * hg;

  for (int kb = 0; kb < K; kb += 32) {
    v8h a0 = *(const v8h*)(arow + kb);
    v8h a1 = *(const v8h*)(arow + kb + 16);
    v16h a;
#pragma unroll
    for (int i = 0; i < 8; ++i) { a[i] = a0[i]; a[i + 8] = a1[i]; }
    v8h b0 = *(const v8h*)(brow + kb);
    v8h b1 = *(const v8h*)(brow + kb + 16);
    v16h b;
#pragma unroll
    for (int i = 0; i < 8; ++i) { b[i] = b0[i]; b[i + 8] = b1[i]; }
    acc = __builtin_amdgcn_wmma_f32_16x16x32_f16(
        /*neg_a=*/false, a, /*neg_b=*/false, b,
        /*c_mod=*/(short)0, acc, /*reuse_a=*/false, /*reuse_b=*/false);
  }
  // C/D layout: VGPR i -> M = i + 8*hg, N = r
  float* crow = C + (size_t)(rowBase + 8 * hg) * Nc + colBase + r;
#pragma unroll
  for (int i = 0; i < 8; ++i) crow[(size_t)i * Nc] = acc[i];
}

// ---------------------------------------------------------------- attention

// h: [n][H][C]; a_s/a_d written as [n*H]
__global__ void k_attn_scores(const float* __restrict__ h,
                              const float* __restrict__ att_s,
                              const float* __restrict__ att_d,
                              float* __restrict__ as_, float* __restrict__ ad_,
                              int n, int H, int C) {
  int i = blockIdx.x * blockDim.x + threadIdx.x;
  if (i >= n * H) return;
  int hd = i % H;
  const float* row = h + (size_t)i * C;
  const float* s = att_s + hd * C;
  const float* d = att_d + hd * C;
  float ss = 0.f, dd = 0.f;
  for (int c = 0; c < C; ++c) { float v = row[c]; ss += v * s[c]; dd += v * d[c]; }
  as_[i] = ss; ad_[i] = dd;
}

__global__ void k_edge_max(const int* __restrict__ esrc, const int* __restrict__ edst,
                           const float* __restrict__ as_, const float* __restrict__ ad_,
                           float* __restrict__ m, int H) {
  int i = blockIdx.x * blockDim.x + threadIdx.x;
  if (i >= NTOT * H) return;
  int hd = i % H, e = i / H;
  int s, d;
  if (e < NEDGES) { s = esrc[e]; d = edst[e]; } else { s = d = e - NEDGES; }
  float l = as_[s * H + hd] + ad_[d * H + hd];
  l = l > 0.f ? l : 0.2f * l;
  atomicMaxFloat(m + d * H + hd, l);
}

// Layer 1: H=2, C=64 (128 features). One wave per edge; lane owns 4 channels.
__global__ void k_edge_accum1(const int* __restrict__ esrc, const int* __restrict__ edst,
                              const float* __restrict__ as_, const float* __restrict__ ad_,
                              const float* __restrict__ m, const float* __restrict__ h,
                              float* __restrict__ den, float* __restrict__ acc) {
  int gid  = blockIdx.x * blockDim.x + threadIdx.x;
  int e    = gid >> 5;
  int lane = gid & 31;
  if (e >= NTOT) return;
  int s, d;
  if (e < NEDGES) { s = esrc[e]; d = edst[e]; } else { s = d = e - NEDGES; }
  int hd = lane >> 4;                         // channel c = lane*4 .. +3 ; head = c/64
  float l = as_[s * 2 + hd] + ad_[d * 2 + hd];
  l = l > 0.f ? l : 0.2f * l;
  float ex = __expf(l - m[d * 2 + hd]);
  if ((lane & 15) == 0) atomicAdd(&den[d * 2 + hd], ex);
  const float4 hv = *(const float4*)(h + (size_t)s * 128 + lane * 4);
  float* o = acc + (size_t)d * 128 + lane * 4;
  atomicAdd(o + 0, ex * hv.x);
  atomicAdd(o + 1, ex * hv.y);
  atomicAdd(o + 2, ex * hv.z);
  atomicAdd(o + 3, ex * hv.w);
}

// Layer 2: H=1, C=64. One wave per edge; lane owns 2 channels.
__global__ void k_edge_accum2(const int* __restrict__ esrc, const int* __restrict__ edst,
                              const float* __restrict__ as_, const float* __restrict__ ad_,
                              const float* __restrict__ m, const float* __restrict__ h,
                              float* __restrict__ den, float* __restrict__ acc) {
  int gid  = blockIdx.x * blockDim.x + threadIdx.x;
  int e    = gid >> 5;
  int lane = gid & 31;
  if (e >= NTOT) return;
  int s, d;
  if (e < NEDGES) { s = esrc[e]; d = edst[e]; } else { s = d = e - NEDGES; }
  float l = as_[s] + ad_[d];
  l = l > 0.f ? l : 0.2f * l;
  float ex = __expf(l - m[d]);
  if (lane == 0) atomicAdd(&den[d], ex);
  const float2 hv = *(const float2*)(h + (size_t)s * 64 + lane * 2);
  float* o = acc + (size_t)d * 64 + lane * 2;
  atomicAdd(o + 0, ex * hv.x);
  atomicAdd(o + 1, ex * hv.y);
}

// finalize layer 1: acc/den + bias, ELU, emit f16 feature matrix for GEMM2
__global__ void k_finalize1(const float* __restrict__ acc, const float* __restrict__ den,
                            const float* __restrict__ b1, _Float16* __restrict__ out_h) {
  int i = blockIdx.x * blockDim.x + threadIdx.x;
  if (i >= NNODES * 128) return;
  int n = i >> 7, c = i & 127, hd = c >> 6;
  float v = acc[i] / den[n * 2 + hd] + b1[c];
  v = v > 0.f ? v : expm1f(v);
  out_h[i] = (_Float16)v;
}

// finalize layer 2 + FC + sigmoid, fused per node
__global__ void k_final(const float* __restrict__ acc, const float* __restrict__ den,
                        const float* __restrict__ b2, const float* __restrict__ fcw,
                        const float* __restrict__ fcb, float* __restrict__ out) {
  int n = blockIdx.x * blockDim.x + threadIdx.x;
  if (n >= NNODES) return;
  float dn = den[n];
  float z = 0.f;
  const float* a = acc + (size_t)n * 64;
  for (int c = 0; c < 64; ++c) {
    float v = a[c] / dn + b2[c];
    v = v > 0.f ? v : expm1f(v);
    z += v * fcw[c];
  }
  z += fcb[0];
  out[n] = 1.f / (1.f + __expf(-z));
}

// ---------------------------------------------------------------- launcher

extern "C" void kernel_launch(void* const* d_in, const int* in_sizes, int n_in,
                              void* d_out, int out_size, void* d_ws, size_t ws_size,
                              hipStream_t stream) {
  const float* x    = (const float*)d_in[0];
  const int*   eidx = (const int*)d_in[1];
  const float* W1   = (const float*)d_in[2];
  const float* at_s1= (const float*)d_in[3];
  const float* at_d1= (const float*)d_in[4];
  const float* b1   = (const float*)d_in[5];
  const float* W2   = (const float*)d_in[6];
  const float* at_s2= (const float*)d_in[7];
  const float* at_d2= (const float*)d_in[8];
  const float* b2   = (const float*)d_in[9];
  const float* fcw  = (const float*)d_in[10];
  const float* fcb  = (const float*)d_in[11];
  float* out = (float*)d_out;

  const int* esrc = eidx;            // edge_index[0]
  const int* edst = eidx + NEDGES;   // edge_index[1]

  char* ws = (char*)d_ws;
  size_t off = 0;
  auto alloc = [&](size_t bytes) -> char* {
    char* p = ws + off;
    off = (off + bytes + 255) & ~(size_t)255;
    return p;
  };
  _Float16* xh   = (_Float16*)alloc((size_t)NNODES * 128 * 2);
  _Float16* w1t  = (_Float16*)alloc(128 * 128 * 2);
  _Float16* w2t  = (_Float16*)alloc(64 * 128 * 2);
  float*    h1   = (float*)   alloc((size_t)NNODES * 128 * 4);
  float*    as1  = (float*)   alloc((size_t)NNODES * 2 * 4);
  float*    ad1  = (float*)   alloc((size_t)NNODES * 2 * 4);
  float*    m1   = (float*)   alloc((size_t)NNODES * 2 * 4);
  float*    den1 = (float*)   alloc((size_t)NNODES * 2 * 4);
  float*    acc1 = (float*)   alloc((size_t)NNODES * 128 * 4);
  _Float16* h1h  = (_Float16*)alloc((size_t)NNODES * 128 * 2);
  float*    h2   = (float*)   alloc((size_t)NNODES * 64 * 4);
  float*    as2  = (float*)   alloc((size_t)NNODES * 4);
  float*    ad2  = (float*)   alloc((size_t)NNODES * 4);
  float*    m2   = (float*)   alloc((size_t)NNODES * 4);
  float*    den2 = (float*)   alloc((size_t)NNODES * 4);
  float*    acc2 = (float*)   alloc((size_t)NNODES * 64 * 4);

  const int T = 256;
  auto g = [](long long n, int t) { return (int)((n + t - 1) / t); };

  // --- prep: f16 copies / transposed weights
  k_f32_to_f16<<<g((long long)NNODES * 128, T), T, 0, stream>>>(x, xh, NNODES * 128);
  k_transpose_f16<<<g(128 * 128, T), T, 0, stream>>>(W1, w1t, 128, 128);
  k_transpose_f16<<<g(128 * 64, T), T, 0, stream>>>(W2, w2t, 128, 64);

  // --- layer 1 GEMM: h1 = x @ W1   (M=50000, K=128, Nc=128; 8 col-tiles/block)
  k_gemm_wmma<<<NNODES / 16, 256, 0, stream>>>(xh, w1t, h1, 128, 128);

  k_attn_scores<<<g((long long)NNODES * 2, T), T, 0, stream>>>(h1, at_s1, at_d1, as1, ad1,
                                                               NNODES, 2, 64);
  k_fill<<<g((long long)NNODES * 2, T), T, 0, stream>>>(m1, -1e30f, NNODES * 2);
  k_fill<<<g((long long)NNODES * 2, T), T, 0, stream>>>(den1, 0.f, NNODES * 2);
  k_fill<<<g((long long)NNODES * 128, T), T, 0, stream>>>(acc1, 0.f, NNODES * 128);

  k_edge_max<<<g((long long)NTOT * 2, T), T, 0, stream>>>(esrc, edst, as1, ad1, m1, 2);
  k_edge_accum1<<<g((long long)NTOT * 32, T), T, 0, stream>>>(esrc, edst, as1, ad1, m1, h1,
                                                              den1, acc1);
  k_finalize1<<<g((long long)NNODES * 128, T), T, 0, stream>>>(acc1, den1, b1, h1h);

  // --- layer 2 GEMM: h2 = h1post @ W2   (M=50000, K=128, Nc=64; 4 col-tiles/block)
  k_gemm_wmma<<<NNODES / 16, 128, 0, stream>>>(h1h, w2t, h2, 128, 64);

  k_attn_scores<<<g(NNODES, T), T, 0, stream>>>(h2, at_s2, at_d2, as2, ad2, NNODES, 1, 64);
  k_fill<<<g(NNODES, T), T, 0, stream>>>(m2, -1e30f, NNODES);
  k_fill<<<g(NNODES, T), T, 0, stream>>>(den2, 0.f, NNODES);
  k_fill<<<g((long long)NNODES * 64, T), T, 0, stream>>>(acc2, 0.f, NNODES * 64);

  k_edge_max<<<g(NTOT, T), T, 0, stream>>>(esrc, edst, as2, ad2, m2, 1);
  k_edge_accum2<<<g((long long)NTOT * 32, T), T, 0, stream>>>(esrc, edst, as2, ad2, m2, h2,
                                                              den2, acc2);
  k_final<<<g(NNODES, T), T, 0, stream>>>(acc2, den2, b2, fcw, fcb, out);
}
